// Model_11759620457151
// MI455X (gfx1250) — matrix operand order repaired and checked by reading, not compiled, not run on hardware
//
#include <hip/hip_runtime.h>
#include <hip/hip_bf16.h>
#include <math.h>

// ---------------------------------------------------------------------------
// Hybrid Mamba/Attention/KAN model for MI455X (gfx1250), wave32 + WMMA bf16.
// Operands are pre-converted to fragment-ready bf16 layouts; GEMM/attention
// read fragments directly from global (L2-resident) with b128 loads.
// ---------------------------------------------------------------------------

typedef __attribute__((ext_vector_type(16))) __bf16 v16bf;
typedef __attribute__((ext_vector_type(8)))  __bf16 v8bf;
typedef __attribute__((ext_vector_type(8)))  float  v8f;

#define DM     512
#define NH     8
#define HE     64
#define DI     1024
#define DS     16
#define DTR    32
#define BB_    4
#define LL_    1024
#define ROWS   (BB_*LL_)

// A-fragment k order per (lane-half): [0..7,16..23] / [8..15,24..31]
// -> storing columns with bits 3<->4 swapped makes the fragment contiguous.
__device__ __forceinline__ int swap34(int c) {
    return (c & 7) | ((c & 8) << 1) | ((c & 16) >> 1);
}
__device__ __forceinline__ int swz32(int c) {    // swizzle within each 32-col group
    return (c & ~31) | swap34(c & 31);
}

__device__ __forceinline__ v16bf load_frag(const __bf16* p) {
    const v8bf* q = (const v8bf*)p;              // 16B-aligned by construction
    v8bf lo = q[0], hi = q[1];
    return __builtin_shufflevector(lo, hi, 0, 1, 2, 3, 4, 5, 6, 7,
                                           8, 9, 10, 11, 12, 13, 14, 15);
}

__device__ __forceinline__ v8bf cvt8(float4 a, float4 b, float s) {
    v8bf r;
    r[0] = (__bf16)(a.x * s); r[1] = (__bf16)(a.y * s);
    r[2] = (__bf16)(a.z * s); r[3] = (__bf16)(a.w * s);
    r[4] = (__bf16)(b.x * s); r[5] = (__bf16)(b.y * s);
    r[6] = (__bf16)(b.z * s); r[7] = (__bf16)(b.w * s);
    return r;
}

__device__ __forceinline__ float sigf(float x) { return 1.0f / (1.0f + __expf(-x)); }
__device__ __forceinline__ float siluf(float x) { return x * sigf(x); }

#define WMMA_BF16(a, b, c) \
    __builtin_amdgcn_wmma_f32_16x16x32_bf16(false, (a), false, (b), (short)0, (c), false, false)

// ---------------------------------------------------------------------------
// Layout-conversion kernels (one pass per matrix, L2-resident afterwards)
// ---------------------------------------------------------------------------
// f32 [M x K] (lda) -> bf16 [M x K]; optional swz32 column swizzle + scale
__global__ void cvt_swz_kernel(const float* __restrict__ A, int lda,
                               __bf16* __restrict__ out, int M, int K,
                               float scale, int swz)
{
    int g = blockIdx.x * 256 + threadIdx.x;      // one 8-column group per thread
    if (g >= M * (K / 8)) return;
    int r  = g / (K / 8);
    int c8 = (g - r * (K / 8)) * 8;
    const float4* src = (const float4*)&A[(size_t)r * lda + c8];
    float4 f0 = src[0], f1 = src[1];
    int cd = swz ? swz32(c8) : c8;
    *(v8bf*)&out[(size_t)r * K + cd] = cvt8(f0, f1, scale);
}

// weights f32 [K x N] -> bf16 transposed [N x K] (B-fragment-ready)
__global__ void wt_kernel(const float* __restrict__ W, __bf16* __restrict__ WT,
                          int K, int N)
{
    int idx = blockIdx.x * 256 + threadIdx.x;
    if (idx >= K * N) return;
    int k = idx / N, n = idx - k * N;            // linear (coalesced) read
    WT[(size_t)n * K + k] = (__bf16)W[idx];
}

// v f32 [B,L,H,E] -> bf16 [B,H,E,L] (B-fragments of P@V contiguous in pos)
__global__ void vt_kernel(const float* __restrict__ v, __bf16* __restrict__ VT)
{
    int idx = blockIdx.x * 256 + threadIdx.x;    // ROWS*DM elements
    if (idx >= ROWS * DM) return;
    int e   = idx & (HE - 1);
    int h   = (idx >> 6) & (NH - 1);
    int pos = (idx >> 9) & (LL_ - 1);
    int b   = idx >> 19;
    VT[(((size_t)(b * NH + h) * HE) + e) * LL_ + swz32(pos)] = (__bf16)v[idx];
}

// ---------------------------------------------------------------------------
// GEMM: C[M,N] = act(Abf[MxK] @ WT[NxK]^T + bias); no LDS, no barriers.
// Block 128x64, 8 waves x (32x32); branch-free 2-stage pipelined loads.
// act: 0 = none, 1 = softplus
// ---------------------------------------------------------------------------
#define GBM 128
#define GBN 64
#define GBK 32

__global__ __launch_bounds__(256) void gemm_wmma_bf16(
    const __bf16* __restrict__ Abf,
    const __bf16* __restrict__ WT,
    const float* __restrict__ bias,
    float* __restrict__ C, int ldc,
    int M, int N, int K, int act)
{
    const int tid  = threadIdx.x;
    const int lane = tid & 31;
    const int wave = tid >> 5;
    const int half = lane >> 4;
    const int m16  = lane & 15;
    const int wm   = wave >> 1;        // 0..3 -> 32-row sub-tile
    const int wn   = wave & 1;         // 0..1 -> 32-col sub-tile
    const int row0 = blockIdx.y * GBM;
    const int col0 = blockIdx.x * GBN;

    const __bf16* arow0 = Abf + (size_t)(row0 + wm * 32 + m16) * K;
    const __bf16* arow1 = arow0 + (size_t)16 * K;
    const __bf16* brow0 = WT + (size_t)(col0 + wn * 32 + m16) * K;
    const __bf16* brow1 = brow0 + (size_t)16 * K;
    const int foff = half << 4;

    v8f acc[4];
    #pragma unroll
    for (int s = 0; s < 4; ++s)
        #pragma unroll
        for (int r = 0; r < 8; ++r) acc[s][r] = 0.0f;

    v16bf a0 = load_frag(arow0 + foff);
    v16bf a1 = load_frag(arow1 + foff);
    v16bf b0 = load_frag(brow0 + foff);
    v16bf b1 = load_frag(brow1 + foff);

    // branch-free pipeline: all-but-last steps load next tile unconditionally
    for (int k0 = 0; k0 + GBK < K; k0 += GBK) {
        int off = k0 + GBK + foff;
        v16bf na0 = load_frag(arow0 + off);
        v16bf na1 = load_frag(arow1 + off);
        v16bf nb0 = load_frag(brow0 + off);
        v16bf nb1 = load_frag(brow1 + off);
        acc[0] = WMMA_BF16(a0, b0, acc[0]);
        acc[1] = WMMA_BF16(a0, b1, acc[1]);
        acc[2] = WMMA_BF16(a1, b0, acc[2]);
        acc[3] = WMMA_BF16(a1, b1, acc[3]);
        a0 = na0; a1 = na1; b0 = nb0; b1 = nb1;
    }
    // peeled final step
    acc[0] = WMMA_BF16(a0, b0, acc[0]);
    acc[1] = WMMA_BF16(a0, b1, acc[1]);
    acc[2] = WMMA_BF16(a1, b0, acc[2]);
    acc[3] = WMMA_BF16(a1, b1, acc[3]);

    #pragma unroll
    for (int ai = 0; ai < 2; ++ai) {
        #pragma unroll
        for (int s = 0; s < 2; ++s) {
            int col = col0 + wn * 32 + s * 16 + m16;
            float bv = bias ? bias[col] : 0.0f;
            #pragma unroll
            for (int r = 0; r < 8; ++r) {
                int row = row0 + wm * 32 + ai * 16 + r + (half << 3);
                float v = acc[ai * 2 + s][r] + bv;
                if (act == 1) v = (v > 20.0f) ? v : __logf(1.0f + __expf(v));
                C[(size_t)row * ldc + col] = v;
            }
        }
    }
}

// ---------------------------------------------------------------------------
// Depthwise causal conv (width 4) + SiLU.  xz is (ROWS, 2*DI); take cols[0:DI]
// ---------------------------------------------------------------------------
__global__ void conv_silu_kernel(const float* __restrict__ xz,
                                 const float* __restrict__ w,
                                 const float* __restrict__ bias,
                                 float* __restrict__ out)
{
    int idx = blockIdx.x * 256 + threadIdx.x;
    if (idx >= ROWS * DI) return;
    int d = idx & (DI - 1);
    int t = (idx >> 10) & (LL_ - 1);
    int b = idx >> 20;
    size_t base = (size_t)b * LL_ * (2 * DI);
    float acc = bias[d];
    #pragma unroll
    for (int k = 0; k < 4; ++k) {
        int tt = t - 3 + k;
        if (tt >= 0) acc += w[d * 4 + k] * xz[base + (size_t)tt * (2 * DI) + d];
    }
    out[(size_t)(b * LL_ + t) * DI + d] = siluf(acc);
}

// ---------------------------------------------------------------------------
// Selective scan. 1 block per batch, 1024 threads (one per inner channel).
// ---------------------------------------------------------------------------
__global__ __launch_bounds__(1024) void scan_kernel(
    const float* __restrict__ dt, const float* __restrict__ xc,
    const float* __restrict__ dbc, const float* __restrict__ xz,
    const float* __restrict__ A_log, const float* __restrict__ Dp,
    float* __restrict__ y)
{
    __shared__ float BC[32];
    const int d = threadIdx.x;
    const int b = blockIdx.x;
    float A[DS], h[DS];
    #pragma unroll
    for (int n = 0; n < DS; ++n) { A[n] = -__expf(A_log[d * DS + n]); h[n] = 0.0f; }
    const float Dd = Dp[d];

    for (int t = 0; t < LL_; ++t) {
        size_t rb = (size_t)(b * LL_ + t);
        if (d < 32) BC[d] = dbc[rb * 64 + 32 + d];   // [0:16]=B_t  [16:32]=C_t
        __syncthreads();
        float dtv = dt[rb * DI + d];
        float xv  = xc[rb * DI + d];
        float zv  = xz[rb * (2 * DI) + DI + d];
        float dx  = dtv * xv;
        float acc = 0.0f;
        #pragma unroll
        for (int n = 0; n < DS; ++n) {
            h[n] = __expf(dtv * A[n]) * h[n] + dx * BC[n];
            acc += h[n] * BC[16 + n];
        }
        acc += xv * Dd;
        y[rb * DI + d] = acc * siluf(zv);
        __syncthreads();
    }
}

// ---------------------------------------------------------------------------
// Residual add + LayerNorm over D=512. One block (128 thr) per row.
// ---------------------------------------------------------------------------
__global__ __launch_bounds__(128) void ln_add_kernel(
    const float* __restrict__ x, const float* __restrict__ r,
    const float* __restrict__ w, const float* __restrict__ bb,
    float* __restrict__ out)
{
    __shared__ float red[128];
    const int row = blockIdx.x, tid = threadIdx.x;
    float v[4]; float s = 0.0f;
    #pragma unroll
    for (int i = 0; i < 4; ++i) {
        int c = tid + i * 128;
        v[i] = x[(size_t)row * DM + c] + r[(size_t)row * DM + c];
        s += v[i];
    }
    red[tid] = s; __syncthreads();
    for (int off = 64; off > 0; off >>= 1) { if (tid < off) red[tid] += red[tid + off]; __syncthreads(); }
    float mean = red[0] * (1.0f / DM);
    __syncthreads();
    float vs = 0.0f;
    #pragma unroll
    for (int i = 0; i < 4; ++i) { float dfl = v[i] - mean; vs += dfl * dfl; }
    red[tid] = vs; __syncthreads();
    for (int off = 64; off > 0; off >>= 1) { if (tid < off) red[tid] += red[tid + off]; __syncthreads(); }
    float inv = rsqrtf(red[0] * (1.0f / DM) + 1e-5f);
    #pragma unroll
    for (int i = 0; i < 4; ++i) {
        int c = tid + i * 128;
        out[(size_t)row * DM + c] = (v[i] - mean) * inv * w[c] + bb[c];
    }
}

// ---------------------------------------------------------------------------
// Flash-style causal attention, barrier-free. grid (L/64, H, B), 128 thr.
// qbf: bf16 swz32 cols (A of QK^T, 1/8 prescaled); kbf: bf16 swz32 cols
// (B of QK^T); vtb: bf16 [b,h,e,pos] (B of P@V). P via wave-private LDS.
// ---------------------------------------------------------------------------
#define TPS 72
__global__ __launch_bounds__(128) void attn_kernel(
    const __bf16* __restrict__ qbf, const __bf16* __restrict__ kbf,
    const __bf16* __restrict__ vtb, float* __restrict__ o)
{
    __shared__ __align__(16) __bf16 Ps[64 * TPS];   // wave-private 16-row slabs
    const int bi = blockIdx.x, h = blockIdx.y, b = blockIdx.z;
    const int tid = threadIdx.x;
    const int lane = tid & 31, wave = tid >> 5;
    const int half = lane >> 4, m16 = lane & 15;
    const int qrow0 = bi * 64;
    const int foff = half << 4;

    const __bf16* qrow = qbf + ((size_t)(b * LL_ + qrow0 + wave * 16 + m16)) * DM + h * HE;
    const __bf16* kbase = kbf + ((size_t)b * LL_) * DM + h * HE;
    const __bf16* vbase = vtb + ((size_t)(b * NH + h) * HE) * LL_;
    __bf16* prow = &Ps[(wave * 16 + m16) * TPS];

    v8f oacc[4]; float m_run[8], l_run[8];
    #pragma unroll
    for (int s = 0; s < 4; ++s)
        #pragma unroll
        for (int rr = 0; rr < 8; ++rr) oacc[s][rr] = 0.0f;
    #pragma unroll
    for (int rr = 0; rr < 8; ++rr) { m_run[rr] = -1e30f; l_run[rr] = 0.0f; }

    for (int jt = 0; jt <= bi; ++jt) {
        const int krow0 = jt * 64;

        // S = Q @ K^T   (wave: 16 rows x 64 cols)
        v8f sacc[4];
        #pragma unroll
        for (int s = 0; s < 4; ++s)
            #pragma unroll
            for (int rr = 0; rr < 8; ++rr) sacc[s][rr] = 0.0f;

        #pragma unroll
        for (int kk = 0; kk < 64; kk += 32) {
            v16bf a = load_frag(qrow + kk + foff);
            #pragma unroll
            for (int s = 0; s < 4; ++s) {
                int n = krow0 + s * 16 + m16;     // key position
                v16bf bf = load_frag(kbase + (size_t)n * DM + kk + foff);
                sacc[s] = WMMA_BF16(a, bf, sacc[s]);
            }
        }

        if (jt == bi) {  // causal mask on diagonal tile
            #pragma unroll
            for (int s = 0; s < 4; ++s) {
                int col = krow0 + s * 16 + m16;
                #pragma unroll
                for (int rr = 0; rr < 8; ++rr) {
                    int row = qrow0 + wave * 16 + rr + (half << 3);
                    if (col > row) sacc[s][rr] = -1e9f;
                }
            }
        }

        // online softmax (row stats across 16-lane group)
        #pragma unroll
        for (int rr = 0; rr < 8; ++rr) {
            float vm = fmaxf(fmaxf(sacc[0][rr], sacc[1][rr]),
                             fmaxf(sacc[2][rr], sacc[3][rr]));
            #pragma unroll
            for (int msk = 1; msk <= 8; msk <<= 1) vm = fmaxf(vm, __shfl_xor(vm, msk, 32));
            float mn = fmaxf(m_run[rr], vm);
            float alpha = __expf(m_run[rr] - mn);
            float rs = 0.0f;
            #pragma unroll
            for (int s = 0; s < 4; ++s) {
                float p = __expf(sacc[s][rr] - mn);
                sacc[s][rr] = p; rs += p;
            }
            #pragma unroll
            for (int msk = 1; msk <= 8; msk <<= 1) rs += __shfl_xor(rs, msk, 32);
            l_run[rr] = l_run[rr] * alpha + rs;
            m_run[rr] = mn;
            #pragma unroll
            for (int s = 0; s < 4; ++s) oacc[s][rr] *= alpha;
        }

        // stash P (bf16, swz32 columns) -- wave-private, no barrier needed
        #pragma unroll
        for (int s = 0; s < 4; ++s)
            #pragma unroll
            for (int rr = 0; rr < 8; ++rr)
                Ps[(wave * 16 + rr + (half << 3)) * TPS + swz32(s * 16 + m16)] =
                    (__bf16)sacc[s][rr];

        // O += P @ V
        #pragma unroll
        for (int kk = 0; kk < 64; kk += 32) {
            v16bf a = load_frag(prow + kk + foff);
            #pragma unroll
            for (int s = 0; s < 4; ++s) {
                int n = s * 16 + m16;             // e index
                v16bf bf = load_frag(vbase + (size_t)n * LL_ + krow0 + kk + foff);
                oacc[s] = WMMA_BF16(a, bf, oacc[s]);
            }
        }
    }

    #pragma unroll
    for (int s = 0; s < 4; ++s) {
        int e = s * 16 + m16;
        #pragma unroll
        for (int rr = 0; rr < 8; ++rr) {
            int row = qrow0 + wave * 16 + rr + (half << 3);
            o[((size_t)(b * LL_ + row)) * DM + h * HE + e] = oacc[s][rr] / l_run[rr];
        }
    }
}

// ---------------------------------------------------------------------------
// KAN head
// ---------------------------------------------------------------------------
__global__ __launch_bounds__(256) void kan_kernel(
    const float* __restrict__ x, const float* __restrict__ base_w,
    const float* __restrict__ scaler, const float* __restrict__ spline_w,
    float* __restrict__ out)
{
    __shared__ float oacc[7];
    const int row = blockIdx.x, tid = threadIdx.x;
    if (tid < 7) oacc[tid] = 0.0f;
    __syncthreads();

    float acc[7];
    #pragma unroll
    for (int c = 0; c < 7; ++c) acc[c] = 0.0f;

    for (int d = tid; d < DM; d += 256) {
        float xv = x[(size_t)row * DM + d];
        float sv = siluf(xv);
        float bs[11];
        #pragma unroll
        for (int i = 0; i < 11; ++i) {
            float g0 = 0.4f * (float)(i - 3) - 1.0f;
            bs[i] = (xv >= g0 && xv < g0 + 0.4f) ? 1.0f : 0.0f;
        }
        #pragma unroll
        for (int j = 1; j <= 3; ++j) {
            float inv = 1.0f / (0.4f * (float)j);
            #pragma unroll
            for (int i = 0; i < 10; ++i) {
                if (i < 11 - j) {
                    float gi = 0.4f * (float)(i - 3) - 1.0f;
                    bs[i] = (xv - gi) * inv * bs[i] +
                            ((gi + 0.4f * (float)(j + 1)) - xv) * inv * bs[i + 1];
                }
            }
        }
        #pragma unroll
        for (int c = 0; c < 7; ++c) {
            float a  = sv * base_w[c * DM + d];
            float sc = scaler[c * DM + d];
            const float* sw = &spline_w[((size_t)c * DM + d) * 8];
            float sp = 0.0f;
            #pragma unroll
            for (int kk = 0; kk < 8; ++kk) sp += bs[kk] * sw[kk];
            acc[c] += a + sc * sp;
        }
    }
    #pragma unroll
    for (int c = 0; c < 7; ++c) {
        float vsum = acc[c];
        #pragma unroll
        for (int msk = 1; msk <= 16; msk <<= 1) vsum += __shfl_xor(vsum, msk, 32);
        if ((tid & 31) == 0) atomicAdd(&oacc[c], vsum);
    }
    __syncthreads();
    if (tid < 7) out[(size_t)row * 7 + tid] = oacc[tid];
}

// ---------------------------------------------------------------------------
// Host orchestration
// ---------------------------------------------------------------------------
extern "C" void kernel_launch(void* const* d_in, const int* in_sizes, int n_in,
                              void* d_out, int out_size, void* d_ws, size_t ws_size,
                              hipStream_t stream)
{
    (void)in_sizes; (void)out_size; (void)ws_size;
    if (n_in < 57) return;
    const float* const* in = (const float* const*)d_in;

    // JAX pytree (sorted-key) flatten order:
    // 0..2   kan: base_w, scaler, spline_w
    // 3+21*l layers[l]: attn{Wk,Wo,Wq,Wv,bk,bo,bq,bv}, ln1{b,w}, ln2{b,w},
    //                   mamba{A_log,D,W_dt,W_in,W_out,W_x,b_dt,conv_b,conv_w}
    // 45..46 ln0: b, w ; 47..55 mamba0 (same key order) ; 56 x
    const float* x_in = in[56];

    char* ws = (char*)d_ws;
    const size_t MB = 1024ull * 1024ull;
    float* xz  = (float*)(ws + 0);           // 32 MB (ROWS x 2048), aliased q/k/v/o
    float* xc  = (float*)(ws + 32 * MB);     // 16 MB
    float* dtb = (float*)(ws + 48 * MB);     // 16 MB
    float* yb  = (float*)(ws + 64 * MB);     // 16 MB (aliased qbf/kbf/vtb in attn)
    float* dbc = (float*)(ws + 80 * MB);     //  1 MB
    float* sub = (float*)(ws + 82 * MB);     //  8 MB
    float* xa  = (float*)(ws + 90 * MB);     //  8 MB
    float* xb  = (float*)(ws + 98 * MB);     //  8 MB
    __bf16* abf = (__bf16*)(ws + 106 * MB);  //  8 MB bf16 A scratch
    __bf16* wtb = (__bf16*)(ws + 114 * MB);  //  2 MB bf16 W^T scratch
    float* qb = xz;
    float* kb = xz + 2ull * 1024 * 1024;
    float* vb = xz + 4ull * 1024 * 1024;
    float* ob = xz + 6ull * 1024 * 1024;
    __bf16* qbf = (__bf16*)yb;               // yb free during attention
    __bf16* kbf = qbf + 2ull * 1024 * 1024;
    __bf16* vtb = kbf + 2ull * 1024 * 1024;

    auto gemm = [&](const float* A, int lda, const float* W,
                    const float* bias, float* C, int ldc, int M, int N, int K, int act) {
        cvt_swz_kernel<<<(M * (K / 8) + 255) / 256, 256, 0, stream>>>(A, lda, abf, M, K, 1.0f, 0);
        wt_kernel<<<(K * N + 255) / 256, 256, 0, stream>>>(W, wtb, K, N);
        gemm_wmma_bf16<<<dim3(N / GBN, M / GBM), 256, 0, stream>>>(
            abf, wtb, bias, C, ldc, M, N, K, act);
    };

    auto mamba = [&](const float* xin,
                     const float* A_log, const float* Dp, const float* W_dt,
                     const float* W_in, const float* W_out, const float* W_x,
                     const float* b_dt, const float* conv_b, const float* conv_w,
                     float* outb) {
        gemm(xin, DM, W_in, nullptr, xz, 2 * DI, ROWS, 2 * DI, DM, 0);
        conv_silu_kernel<<<(ROWS * DI + 255) / 256, 256, 0, stream>>>(xz, conv_w, conv_b, xc);
        gemm(xc, DI, W_x, nullptr, dbc, 64, ROWS, 64, DI, 0);
        gemm(dbc, 64, W_dt, b_dt, dtb, DI, ROWS, DI, DTR, 1);
        scan_kernel<<<BB_, 1024, 0, stream>>>(dtb, xc, dbc, xz, A_log, Dp, yb);
        gemm(yb, DI, W_out, nullptr, outb, DM, ROWS, DM, DI, 0);
    };

    // --- mamba0 + ln0 ---
    mamba(x_in, in[47], in[48], in[49], in[50], in[51], in[52], in[53], in[54], in[55], sub);
    ln_add_kernel<<<ROWS, 128, 0, stream>>>(x_in, sub, in[46], in[45], xa);

    // --- layers ---
    for (int l = 0; l < 2; ++l) {
        const int p = 3 + 21 * l;
        const float *Wk = in[p+0], *Wo = in[p+1], *Wq = in[p+2], *Wv = in[p+3];
        const float *bk = in[p+4], *bo = in[p+5], *bq = in[p+6], *bv = in[p+7];
        const float *ln1b = in[p+8],  *ln1w = in[p+9];
        const float *ln2b = in[p+10], *ln2w = in[p+11];

        gemm(xa, DM, Wq, bq, qb, DM, ROWS, DM, DM, 0);
        gemm(xa, DM, Wk, bk, kb, DM, ROWS, DM, DM, 0);
        gemm(xa, DM, Wv, bv, vb, DM, ROWS, DM, DM, 0);

        cvt_swz_kernel<<<(ROWS * (DM / 8) + 255) / 256, 256, 0, stream>>>(qb, DM, qbf, ROWS, DM, 0.125f, 1);
        cvt_swz_kernel<<<(ROWS * (DM / 8) + 255) / 256, 256, 0, stream>>>(kb, DM, kbf, ROWS, DM, 1.0f, 1);
        vt_kernel<<<(ROWS * DM + 255) / 256, 256, 0, stream>>>(vb, vtb);
        attn_kernel<<<dim3(LL_ / 64, NH, BB_), 128, 0, stream>>>(qbf, kbf, vtb, ob);

        gemm(ob, DM, Wo, bo, sub, DM, ROWS, DM, DM, 0);
        ln_add_kernel<<<ROWS, 128, 0, stream>>>(xa, sub, ln1w, ln1b, xb);

        mamba(xb, in[p+12], in[p+13], in[p+14], in[p+15], in[p+16],
              in[p+17], in[p+18], in[p+19], in[p+20], sub);
        ln_add_kernel<<<ROWS, 128, 0, stream>>>(xb, sub, ln2w, ln2b, xa);
    }

    // --- KAN head ---
    kan_kernel<<<ROWS, 256, 0, stream>>>(xa, in[0], in[1], in[2], (float*)d_out);
}
